// CrossAttention_29532195127496
// MI455X (gfx1250) — matrix operand order repaired and checked by reading, not compiled, not run on hardware
//
#include <hip/hip_runtime.h>
#include <hip/hip_bf16.h>

// Problem constants
#define B_    16
#define SKV_  4096
#define D_    1024
#define H_    16
#define HD_   64
static constexpr float SCALE = 0.03125f; // 1/sqrt(1024)

typedef __attribute__((ext_vector_type(16))) __bf16        v16bf;
typedef __attribute__((ext_vector_type(8)))  float         v8f;
typedef __attribute__((ext_vector_type(4)))  unsigned int  u32x4;
typedef __attribute__((ext_vector_type(8)))  unsigned int  u32x8;

__device__ __forceinline__ unsigned short f2bf(float f) {
    unsigned int u = __float_as_uint(f);
    u += 0x7FFFu + ((u >> 16) & 1u);           // round-to-nearest-even
    return (unsigned short)(u >> 16);
}
__device__ __forceinline__ unsigned int pkbf(float lo, float hi) {
    return (unsigned int)f2bf(lo) | ((unsigned int)f2bf(hi) << 16);
}

// ---------------- Stage A / F / G: simple row-GEMV projections ----------------
// Y[b][o] = X[b][:] . W[o][:] + bias[o]        (B_=16 rows, 1024x1024 weight)
__global__ void __launch_bounds__(256) gemv_full(
    const float* __restrict__ X, const float* __restrict__ W,
    const float* __restrict__ bias, float* __restrict__ Y) {
    const int o = blockIdx.x * 256 + threadIdx.x;
    const int b = blockIdx.y;
    const float4* x4 = (const float4*)(X + (size_t)b * 1024);
    const float4* w4 = (const float4*)(W + (size_t)o * 1024);
    float acc = 0.f;
#pragma unroll 4
    for (int i = 0; i < 256; ++i) {
        float4 a = x4[i], c = w4[i];
        acc = fmaf(a.x, c.x, acc); acc = fmaf(a.y, c.y, acc);
        acc = fmaf(a.z, c.z, acc); acc = fmaf(a.w, c.w, acc);
    }
    Y[(size_t)b * 1024 + o] = acc + bias[o];
}

// attn[b][o] = wv[b][o>>6][:] . Wv[o][:] + bv[o]   (per-head input row)
__global__ void __launch_bounds__(256) gemv_head(
    const float* __restrict__ WV, const float* __restrict__ W,
    const float* __restrict__ bias, float* __restrict__ Y) {
    const int o = blockIdx.x * 256 + threadIdx.x;
    const int b = blockIdx.y;
    const float4* x4 = (const float4*)(WV + ((size_t)(b * H_ + (o >> 6))) * 1024);
    const float4* w4 = (const float4*)(W + (size_t)o * 1024);
    float acc = 0.f;
#pragma unroll 4
    for (int i = 0; i < 256; ++i) {
        float4 a = x4[i], c = w4[i];
        acc = fmaf(a.x, c.x, acc); acc = fmaf(a.y, c.y, acc);
        acc = fmaf(a.z, c.z, acc); acc = fmaf(a.w, c.w, acc);
    }
    Y[(size_t)b * 1024 + o] = acc + bias[o];
}

// ---------------- Stage B: fold q into Wk -> qk_eff (bf16), and cb ----------------
// qk_eff[b][h][i] = sum_d qh[b][h*64+d] * Wk[h*64+d][i]
__global__ void __launch_bounds__(256) qk_fold(
    const float* __restrict__ qh, const float* __restrict__ Wk,
    unsigned short* __restrict__ qkbf) {
    const int i = blockIdx.x * 256 + threadIdx.x;
    const int h = blockIdx.y, b = blockIdx.z;
    const float* q = qh + (size_t)b * 1024 + h * HD_;
    float acc = 0.f;
#pragma unroll 8
    for (int d = 0; d < HD_; ++d)
        acc = fmaf(q[d], Wk[(size_t)(h * HD_ + d) * 1024 + i], acc);
    qkbf[((size_t)(b * H_ + h)) * 1024 + i] = f2bf(acc);
}

// cb[b][h] = qh[b][h*64:...] . bk[h*64:...]
__global__ void cb_kern(const float* __restrict__ qh, const float* __restrict__ bk,
                        float* __restrict__ cb) {
    const int t = threadIdx.x;            // 256 = (b,h)
    const int b = t >> 4, h = t & 15;
    float acc = 0.f;
    for (int d = 0; d < HD_; ++d)
        acc = fmaf(qh[(size_t)b * 1024 + h * HD_ + d], bk[h * HD_ + d], acc);
    cb[t] = acc;
}

// ---------------- Stage C: scores[b][h][s] via WMMA bf16 ----------------
// D(16h x 16s) += A(16h x 32k of qk_eff) x B(32k x 16s of key^T); K=1024
__global__ void __launch_bounds__(256) scores_wmma(
    const unsigned short* __restrict__ qkbf, const float* __restrict__ key,
    const float* __restrict__ cb, float* __restrict__ scores) {
    const int lane = threadIdx.x & 31;
    const int wid  = threadIdx.x >> 5;
    const int b  = blockIdx.y;
    const int s0 = (blockIdx.x * 8 + wid) * 16;
    const int m  = lane & 15;
    const bool hi = lane >= 16;

    const unsigned short* arow = qkbf + ((size_t)(b * H_ + m)) * 1024;
    const float* brow = key + ((size_t)b * SKV_ + s0 + m) * 1024 + (hi ? 16 : 0);

    v8f acc = {};
    for (int k0 = 0; k0 < 1024; k0 += 32) {
        // A fragment (contiguous K: element e of lane-half h -> K k0+16h+e)
        u32x8 abits;
        *((u32x4*)&abits)     = *(const u32x4*)(arow + k0 + (hi ? 16 : 0));
        *((u32x4*)&abits + 1) = *(const u32x4*)(arow + k0 + (hi ? 24 : 8));

        // B fragment: lane n holds column s0+n, K contiguous per lane-half -> 16 consecutive floats
        const float4* kp = (const float4*)(brow + k0);
        float4 f0 = kp[0], f1 = kp[1], f2 = kp[2], f3 = kp[3];
        u32x8 bbits;
        bbits[0] = pkbf(f0.x, f0.y); bbits[1] = pkbf(f0.z, f0.w);
        bbits[2] = pkbf(f1.x, f1.y); bbits[3] = pkbf(f1.z, f1.w);
        bbits[4] = pkbf(f2.x, f2.y); bbits[5] = pkbf(f2.z, f2.w);
        bbits[6] = pkbf(f3.x, f3.y); bbits[7] = pkbf(f3.z, f3.w);

        acc = __builtin_amdgcn_wmma_f32_16x16x32_bf16(
            false, __builtin_bit_cast(v16bf, abits),
            false, __builtin_bit_cast(v16bf, bbits),
            (short)0, acc, false, false);
    }
    // D layout: lane -> N (=s), VGPR r -> M = r + 8*(lane>=16) (=h)
    const int s  = s0 + m;
    const int hb = hi ? 8 : 0;
#pragma unroll
    for (int r = 0; r < 8; ++r) {
        const int h = hb + r;
        scores[((size_t)(b * H_ + h)) * SKV_ + s] = (acc[r] + cb[b * H_ + h]) * SCALE;
    }
}

// ---------------- Stage D: softmax over s (fp32 in, bf16 weights out) ----------------
__global__ void __launch_bounds__(256) softmax_k(
    const float* __restrict__ scores, unsigned short* __restrict__ wbf) {
    const int row = blockIdx.x;                   // b*16+h
    const float* s = scores + (size_t)row * SKV_;
    unsigned short* w = wbf + (size_t)row * SKV_;
    __shared__ float red[256];
    const int t = threadIdx.x;

    float mx = -3.4e38f;
    for (int i = t; i < SKV_; i += 256) mx = fmaxf(mx, s[i]);
    red[t] = mx; __syncthreads();
    for (int o = 128; o > 0; o >>= 1) { if (t < o) red[t] = fmaxf(red[t], red[t + o]); __syncthreads(); }
    mx = red[0]; __syncthreads();

    float sum = 0.f;
    for (int i = t; i < SKV_; i += 256) sum += __expf(s[i] - mx);
    red[t] = sum; __syncthreads();
    for (int o = 128; o > 0; o >>= 1) { if (t < o) red[t] += red[t + o]; __syncthreads(); }
    const float inv = 1.0f / red[0];

    for (int i = t; i < SKV_; i += 256) w[i] = f2bf(__expf(s[i] - mx) * inv);
}

// ---------------- Stage E: wv[b][h][i] = sum_s w[b][h][s]*value[b][s][i] via WMMA ----------------
// One wave per (b, 16-wide i-tile); K=4096 over s; value transposed through LDS.
__global__ void __launch_bounds__(32) wv_wmma(
    const unsigned short* __restrict__ wbf, const float* __restrict__ value,
    float* __restrict__ wv) {
    __shared__ __align__(32) unsigned short tile[32][16];   // 32 s-rows x 16 i-cols (bf16)
    const int lane = threadIdx.x;
    const int b  = blockIdx.y;
    const int i0 = blockIdx.x * 16;
    const int m  = lane & 15;
    const bool hi = lane >= 16;
    const unsigned short* arow = wbf + ((size_t)(b * H_ + m)) * SKV_;

    v8f acc = {};
    for (int sb = 0; sb < SKV_; sb += 32) {
        // stage value[sb+lane][i0..i0+15] into LDS as bf16
        const float4* vp = (const float4*)(value + ((size_t)b * SKV_ + sb + lane) * 1024 + i0);
        float4 f0 = vp[0], f1 = vp[1], f2 = vp[2], f3 = vp[3];
        u32x8 row;
        row[0] = pkbf(f0.x, f0.y); row[1] = pkbf(f0.z, f0.w);
        row[2] = pkbf(f1.x, f1.y); row[3] = pkbf(f1.z, f1.w);
        row[4] = pkbf(f2.x, f2.y); row[5] = pkbf(f2.z, f2.w);
        row[6] = pkbf(f3.x, f3.y); row[7] = pkbf(f3.z, f3.w);
        *(u32x8*)&tile[lane][0] = row;
        __syncthreads();

        // A fragment from contiguous bf16 softmax-weight row
        u32x8 abits;
        *((u32x4*)&abits)     = *(const u32x4*)(arow + sb + (hi ? 16 : 0));
        *((u32x4*)&abits + 1) = *(const u32x4*)(arow + sb + 8 + (hi ? 16 : 0));

        // B fragment: lane n = column i0+n, K = s within tile (contiguous per lane-half)
        const int kb = hi ? 16 : 0;
        u32x8 bbits;
#pragma unroll
        for (int j = 0; j < 8; ++j)
            bbits[j] = (unsigned int)tile[kb + 2 * j][m] |
                       ((unsigned int)tile[kb + 2 * j + 1][m] << 16);

        acc = __builtin_amdgcn_wmma_f32_16x16x32_bf16(
            false, __builtin_bit_cast(v16bf, abits),
            false, __builtin_bit_cast(v16bf, bbits),
            (short)0, acc, false, false);
        __syncthreads();
    }
    const int hb = hi ? 8 : 0;
#pragma unroll
    for (int r = 0; r < 8; ++r)
        wv[((size_t)(b * H_ + hb + r)) * 1024 + i0 + m] = acc[r];
}

// ---------------- Host launcher ----------------
extern "C" void kernel_launch(void* const* d_in, const int* in_sizes, int n_in,
                              void* d_out, int out_size, void* d_ws, size_t ws_size,
                              hipStream_t stream) {
    (void)in_sizes; (void)n_in; (void)out_size; (void)ws_size;
    const float* query = (const float*)d_in[0];
    const float* key   = (const float*)d_in[1];
    const float* value = (const float*)d_in[2];
    const float* Wq = (const float*)d_in[3];
    const float* bq = (const float*)d_in[4];
    const float* Wk = (const float*)d_in[5];
    const float* bk = (const float*)d_in[6];
    const float* Wv = (const float*)d_in[7];
    const float* bv = (const float*)d_in[8];
    const float* Wo = (const float*)d_in[9];
    const float* bo = (const float*)d_in[10];

    char* ws = (char*)d_ws;                                  // ~8.0 MB used
    float*          qh     = (float*)(ws + 0);               // 64 KB  [16][1024]
    float*          attn   = (float*)(ws + 65536);           // 64 KB  [16][1024]
    float*          cb     = (float*)(ws + 131072);          // 1 KB   [16][16]
    float*          wv     = (float*)(ws + 196608);          // 1 MB   [16][16][1024]
    unsigned short* qkbf   = (unsigned short*)(ws + 1310720);// 512 KB [16][16][1024] bf16
    unsigned short* wbf    = (unsigned short*)(ws + 2097152);// 2 MB   [16][16][4096] bf16
    float*          scores = (float*)(ws + 4194304);         // 4 MB   [16][16][4096]
    float*          out    = (float*)d_out;

    gemv_full  <<<dim3(4, 16),      256, 0, stream>>>(query, Wq, bq, qh);
    qk_fold    <<<dim3(4, 16, 16),  256, 0, stream>>>(qh, Wk, qkbf);
    cb_kern    <<<1,                256, 0, stream>>>(qh, bk, cb);
    scores_wmma<<<dim3(32, 16),     256, 0, stream>>>(qkbf, key, cb, scores);
    softmax_k  <<<256,              256, 0, stream>>>(scores, wbf);
    wv_wmma    <<<dim3(64, 16),      32, 0, stream>>>(wbf, value, wv);
    gemv_head  <<<dim3(4, 16),      256, 0, stream>>>(wv, Wv, bv, attn);
    gemv_full  <<<dim3(4, 16),      256, 0, stream>>>(attn, Wo, bo, out);
}
